// Network_10213432230457
// MI455X (gfx1250) — matrix operand order repaired and checked by reading, not compiled, not run on hardware
//
#include <hip/hip_runtime.h>
#include <hip/hip_bf16.h>

typedef _Float16 v8h  __attribute__((ext_vector_type(8)));
typedef _Float16 v16h __attribute__((ext_vector_type(16)));
typedef float    v8f  __attribute__((ext_vector_type(8)));

#define M_TILE    128
#define NTHREADS  512
#define K1        350
#define K1P       352
#define HA_STRIDE 360        // padded: 180 dwords/row -> conflict-free b128 reads
#define N1        256
#define H1_STRIDE 264        // padded: 132 dwords/row -> conflict-free b128 reads
#define NCHUNK    11         // ceil(350/32)
#define CHUNK_E   8192       // f16 elements per K-chunk image: 256 cols x 32 k
#define W1H_E     (NCHUNK * CHUNK_E)   // 90112 f16
#define W2H_E     (16 * 256)           // 4096 f16 (rows 5..15 zero)

#define SHUF16(lo, hi) __builtin_shufflevector((lo), (hi), 0,1,2,3,4,5,6,7,8,9,10,11,12,13,14,15)

// ---------------- pre-pass: convert weights to f16 tile images ----------------
__global__ __launch_bounds__(256) void preconvert_kernel(
    const float* __restrict__ w1, const float* __restrict__ w2,
    _Float16* __restrict__ w1h, _Float16* __restrict__ w2h)
{
    const int idx = blockIdx.x * 256 + threadIdx.x;
    if (idx < W1H_E) {
        // chunk-major: w1h[kc*8192 + n*32 + kk] = w1[n][kc*32+kk], zero-padded past K=350
        const int kc  = idx >> 13;
        const int rem = idx & (CHUNK_E - 1);
        const int n   = rem >> 5;
        const int kk  = rem & 31;
        const int k   = kc * 32 + kk;
        w1h[idx] = (_Float16)((k < K1) ? w1[n * K1 + k] : 0.f);
    } else {
        const int j = idx - W1H_E;
        if (j < W2H_E) {
            const int n = j >> 8;
            const int k = j & 255;
            w2h[j] = (_Float16)((n < 5) ? w2[n * 256 + k] : 0.f);
        }
    }
}

// ---------------- main fused actor-critic kernel ----------------
__launch_bounds__(NTHREADS)
__global__ void actor_critic_kernel(
    const float* __restrict__ state,
    const float* __restrict__ wa,  const float* __restrict__ ba,
    const _Float16* __restrict__ w1h, const float* __restrict__ b1,
    const _Float16* __restrict__ w2h, const float* __restrict__ b2,
    const float* __restrict__ wc,  const float* __restrict__ bc,
    const float* __restrict__ wlc, const float* __restrict__ blc,
    float* __restrict__ out, int batch)
{
    // LDS map (142848 B total, 2 workgroups / 320KB WGP = 8 waves/SIMD):
    //   [0,      17920)  sState : 128 x 35 f32        (later aliased by sLogits 2560 B)
    //   [17920, 110080)  sHa    : 128 x 360 f16       (later aliased by sH1 128 x 264 f16)
    //   [110080,126464)  sB0    : weight chunk buf 0  (256 x 32 f16, XOR-swizzled blocks)
    //   [126464,142848)  sB1    : weight chunk buf 1
    __shared__ __attribute__((aligned(16))) unsigned char smem[142848];
    float*    sState  = (float*)smem;
    float*    sLogits = (float*)smem;                       // alias (post-GEMM2)
    _Float16* sHa     = (_Float16*)(smem + 17920);
    _Float16* sH1     = (_Float16*)(smem + 17920);          // alias (post K-loop)
    _Float16* sB0     = (_Float16*)(smem + 110080);
    _Float16* sB1     = (_Float16*)(smem + 126464);

    const int tid   = threadIdx.x;
    const int lane  = tid & 31;
    const int wave  = tid >> 5;          // 0..15
    const int msub  = wave >> 1;         // 0..7 : 16-row subtile
    const int nseg  = wave & 1;          // 0..1 : 128-col segment
    const int nbase = nseg * 128;
    const int mbase = blockIdx.x * M_TILE;

    // chunk staging mapping: thread covers 16 f16 of row n, with XOR block swizzle
    const int stg_n    = tid >> 1;
    const int stg_b    = (tid & 1) * 16;
    const int stg_swz  = ((stg_n >> 2) & 3) * 8;

    // ---- stage state tile (coalesced) + first weight chunk ----
    for (int idx = tid; idx < M_TILE * 35; idx += NTHREADS)
        sState[idx] = state[mbase * 35 + idx];
    {
        const v8h* src = (const v8h*)(w1h + stg_n * 32 + stg_b);
        v8h r0 = src[0], r1 = src[1];
        *(v8h*)(sB0 + stg_n * 32 + ( stg_b      ^ stg_swz)) = r0;
        *(v8h*)(sB0 + stg_n * 32 + ((stg_b + 8) ^ stg_swz)) = r1;
    }
    __syncthreads();

    // ---- critic: value[m] = blc + sum_{c,p} wlc[c*35+p]*relu(wc[c]*s+bc[c]) ----
    if (tid < M_TILE) {
        float acc = blc[0];
        const float* srow = &sState[tid * 35];
        #pragma unroll
        for (int c = 0; c < 3; ++c) {
            const float wcc = wc[c], bcc = bc[c];
            #pragma unroll
            for (int p = 0; p < 35; ++p) {
                float h = fmaxf(fmaf(wcc, srow[p], bcc), 0.f);
                acc = fmaf(wlc[c * 35 + p], h, acc);
            }
        }
        out[(size_t)batch * 5 + mbase + tid] = acc;
    }

    // ---- build ha tile: ha[mm][k] = relu(wa[k/35]*s[mm][k%35]+ba[k/35]), f16, packed stores ----
    {
        const int mm = tid >> 2;            // 128 rows, 4 threads/row, 88 k's each
        const int q  = tid & 3;
        const float* srow = &sState[mm * 35];
        int k = q * 88;
        int c = k / 35;
        int p = k - c * 35;
        float wac = wa[c], bac = ba[c];
        _Float16* dst = &sHa[mm * HA_STRIDE + k];
        for (int i0 = 0; i0 < 88; i0 += 8) {
            v8h pack;
            #pragma unroll
            for (int j = 0; j < 8; ++j) {
                float v = 0.f;
                if (k + i0 + j < K1)
                    v = fmaxf(fmaf(wac, srow[p], bac), 0.f);
                pack[j] = (_Float16)v;
                if (++p == 35) {
                    p = 0; ++c;
                    wac = (c < 10) ? wa[c] : 0.f;
                    bac = (c < 10) ? ba[c] : 0.f;
                }
            }
            *(v8h*)(dst + i0) = pack;
        }
    }

    // ---- accumulators preloaded with b1 bias ----
    v8f acc[8];
    #pragma unroll
    for (int t = 0; t < 8; ++t) {
        const float bv = b1[nbase + t * 16 + (lane & 15)];
        #pragma unroll
        for (int r = 0; r < 8; ++r) acc[t][r] = bv;
    }

    const int rowA  = msub * 16 + (lane & 15);  // A-frag: lane -> row m
    const int kselA = 16 * (lane >> 4);         // A-frag K: lane half h holds k 16h..16h+15 (contiguous)
    const int kselB = 16 * (lane >> 4);         // B-frag K: lane half h holds k 16h..16h+15 (contiguous)

    __syncthreads();   // sHa + sB0 ready

    // ---- main GEMM: h1 = relu(ha @ w1^T + b1); double-buffered, swizzled weight chunks ----
    for (int kc = 0; kc < NCHUNK; ++kc) {
        _Float16* cur = (kc & 1) ? sB1 : sB0;
        _Float16* nxt = (kc & 1) ? sB0 : sB1;

        if (kc + 1 < NCHUNK) {
            const v8h* src = (const v8h*)(w1h + (kc + 1) * CHUNK_E + stg_n * 32 + stg_b);
            v8h r0 = src[0], r1 = src[1];
            *(v8h*)(nxt + stg_n * 32 + ( stg_b      ^ stg_swz)) = r0;
            *(v8h*)(nxt + stg_n * 32 + ((stg_b + 8) ^ stg_swz)) = r1;
        }
        if (kc + 2 < NCHUNK)
            __builtin_prefetch(w1h + (kc + 2) * CHUNK_E + tid * 16, 0, 3);

        // A fragment: two 16B runs (ds_load_b128 x2), padded stride -> conflict-free
        const _Float16* ap = &sHa[rowA * HA_STRIDE + kc * 32 + kselA];
        v8h alo = *(const v8h*)(ap);
        v8h ahi = *(const v8h*)(ap + 8);
        v16h a = SHUF16(alo, ahi);

        #pragma unroll
        for (int t = 0; t < 8; ++t) {
            const int n   = nbase + t * 16 + (lane & 15);
            const int swz = ((n >> 2) & 3) * 8;
            const _Float16* rowp = cur + n * 32;
            v8h blo = *(const v8h*)(rowp + ( kselB      ^ swz));
            v8h bhi = *(const v8h*)(rowp + ((kselB + 8) ^ swz));
            v16h bm = SHUF16(blo, bhi);
            acc[t] = __builtin_amdgcn_wmma_f32_16x16x32_f16(
                false, a, false, bm, (short)0, acc[t], false, false);
        }
        __syncthreads();
    }

    // ---- write h1 tile (relu, f16) into LDS (aliases dead sHa) ----
    #pragma unroll
    for (int t = 0; t < 8; ++t) {
        const int j = nbase + t * 16 + (lane & 15);
        #pragma unroll
        for (int r = 0; r < 8; ++r) {
            const int mm = msub * 16 + r + 8 * (lane >> 4);
            sH1[mm * H1_STRIDE + j] = (_Float16)fmaxf(acc[t][r], 0.f);
        }
    }
    __syncthreads();

    // ---- logits = h1 @ w2^T + b2 : K=256, N=16 (5 live cols); 8 waves, one per M-subtile ----
    if (nseg == 0) {
        const int n = lane & 15;
        v8f c2;
        const float bias2 = (n < 5) ? b2[n] : 0.f;
        #pragma unroll
        for (int r = 0; r < 8; ++r) c2[r] = bias2;

        #pragma unroll
        for (int kc = 0; kc < 8; ++kc) {
            const _Float16* ap = &sH1[rowA * H1_STRIDE + kc * 32 + kselA];
            v8h alo = *(const v8h*)(ap);
            v8h ahi = *(const v8h*)(ap + 8);
            v16h a = SHUF16(alo, ahi);

            // B from pre-padded f16 w2 image: branch-free 2 x global b128
            const _Float16* bp = w2h + n * 256 + kc * 32 + kselB;
            v8h blo = *(const v8h*)(bp);
            v8h bhi = *(const v8h*)(bp + 8);
            v16h bm = SHUF16(blo, bhi);

            c2 = __builtin_amdgcn_wmma_f32_16x16x32_f16(
                false, a, false, bm, (short)0, c2, false, false);
        }

        if (n < 5) {
            const int mrow = msub * 16 + 8 * (lane >> 4);
            #pragma unroll
            for (int r = 0; r < 8; ++r)
                sLogits[(mrow + r) * 5 + n] = c2[r];
        }
    }
    __syncthreads();

    // ---- coalesced logits writeback: 128 rows x 5 cols ----
    for (int idx = tid; idx < M_TILE * 5; idx += NTHREADS)
        out[(size_t)mbase * 5 + idx] = sLogits[idx];
}

extern "C" void kernel_launch(void* const* d_in, const int* in_sizes, int n_in,
                              void* d_out, int out_size, void* d_ws, size_t ws_size,
                              hipStream_t stream) {
    const float* state = (const float*)d_in[0];
    const float* wa    = (const float*)d_in[1];
    const float* ba    = (const float*)d_in[2];
    const float* w1    = (const float*)d_in[3];
    const float* b1    = (const float*)d_in[4];
    const float* w2    = (const float*)d_in[5];
    const float* b2    = (const float*)d_in[6];
    const float* wc    = (const float*)d_in[7];
    const float* bc    = (const float*)d_in[8];
    const float* wlc   = (const float*)d_in[9];
    const float* blc   = (const float*)d_in[10];
    float* out = (float*)d_out;

    _Float16* w1h = (_Float16*)d_ws;              // 90112 f16 = 180224 B
    _Float16* w2h = w1h + W1H_E;                  // 4096 f16  = 8192 B

    const int batch = in_sizes[0] / 35;           // 262144
    const int total_cvt = W1H_E + W2H_E;          // 94208
    preconvert_kernel<<<(total_cvt + 255) / 256, 256, 0, stream>>>(w1, w2, w1h, w2h);

    const int nblocks = batch / M_TILE;           // 2048 workgroups of 512 threads (16 waves)
    actor_critic_kernel<<<nblocks, NTHREADS, 0, stream>>>(
        state, wa, ba, w1h, b1, w2h, b2, wc, bc, wlc, blc, out, batch);
}
